// GRUCell_37014028156993
// MI455X (gfx1250) — matrix operand order repaired and checked by reading, not compiled, not run on hardware
//
#include <hip/hip_runtime.h>
#include <hip/hip_bf16.h>
#include <stdint.h>
#include <stddef.h>

// ---------------------------------------------------------------------------
// Hyperbolic GRU cell (Poincare ball, c=1) for MI455X / gfx1250.
// B=4096, D_IN=D_OUT=2048.  6 bf16 WMMA GEMMs + rowwise manifold pointwise.
// Data movement: GLOBAL_LOAD_ASYNC_TO_LDS_B128 (ASYNCcnt-pipelined) when the
// toolchain exposes the gfx1250 async builtins; VGPR staging otherwise.
// ---------------------------------------------------------------------------

#define BROWS 4096
#define DIM   2048
#define EPSV  1e-5f
#define MAXNV (1.0f - 1e-5f)

typedef __attribute__((ext_vector_type(16))) __bf16 v16bf;
typedef __attribute__((ext_vector_type(8)))  float  v8f;
typedef int v4i __attribute__((vector_size(16)));

#if defined(__AMDGCN__) && __has_builtin(__builtin_amdgcn_global_load_async_to_lds_b128) && __has_builtin(__builtin_amdgcn_s_wait_asynccnt)
#define USE_ASYNC_LDS 1
#else
#define USE_ASYNC_LDS 0
#endif

__device__ __forceinline__ unsigned short f2bf(float f) {
  unsigned int u = __float_as_uint(f);
  u += 0x7FFFu + ((u >> 16) & 1u);          // round-to-nearest-even
  return (unsigned short)(u >> 16);
}

// wave32 shuffle reduction + one LDS combine (2 barriers total)
__device__ __forceinline__ float blockReduceSum(float v, float* sd) {
  #pragma unroll
  for (int m = 16; m > 0; m >>= 1) v += __shfl_xor(v, m, 32);
  const int tid  = threadIdx.x;
  const int lane = tid & 31;
  const int wv   = tid >> 5;
  if (lane == 0) sd[wv] = v;
  __syncthreads();
  float r = sd[0] + sd[1] + sd[2] + sd[3] + sd[4] + sd[5] + sd[6] + sd[7];
  __syncthreads();
  return r;
}

__device__ __forceinline__ float artanh_c(float n) {
  return 0.5f * log1pf(2.0f * n / (1.0f - n));   // n pre-clipped <= MAXNV
}

__device__ __forceinline__ void zero_exp8(float (&v)[8], float* sd) {
  float ssq = 0.f;
  #pragma unroll
  for (int j = 0; j < 8; ++j) ssq += v[j] * v[j];
  ssq = blockReduceSum(ssq, sd);
  float n = fmaxf(sqrtf(ssq), EPSV);
  float s = tanhf(n) / n;
  #pragma unroll
  for (int j = 0; j < 8; ++j) v[j] *= s;
}

__device__ __forceinline__ void zero_log8(float (&v)[8], float* sd) {
  float ssq = 0.f;
  #pragma unroll
  for (int j = 0; j < 8; ++j) ssq += v[j] * v[j];
  ssq = blockReduceSum(ssq, sd);
  float n = fminf(fmaxf(sqrtf(ssq), EPSV), MAXNV);
  float s = artanh_c(n) / n;
  #pragma unroll
  for (int j = 0; j < 8; ++j) v[j] *= s;
}

__device__ __forceinline__ void mobius8(float (&x)[8], const float (&y)[8], float* sd) {
  float xy = 0.f, x2 = 0.f, y2 = 0.f;
  #pragma unroll
  for (int j = 0; j < 8; ++j) { xy += x[j]*y[j]; x2 += x[j]*x[j]; y2 += y[j]*y[j]; }
  xy = blockReduceSum(xy, sd);
  x2 = blockReduceSum(x2, sd);
  y2 = blockReduceSum(y2, sd);
  float cA = 1.f + 2.f*xy + y2;
  float cB = 1.f - x2;
  float inv = 1.f / fmaxf(1.f + 2.f*xy + x2*y2, EPSV);
  #pragma unroll
  for (int j = 0; j < 8; ++j) x[j] = (cA*x[j] + cB*y[j]) * inv;
}

// ---------------------------------------------------------------------------
__global__ __launch_bounds__(256) void zero_log_prep(
    const float* __restrict__ in, unsigned short* __restrict__ out_bf,
    float* __restrict__ out_f) {
  __shared__ float sd[8];
  const int tid = threadIdx.x;
  const size_t base = (size_t)blockIdx.x * DIM;
  float x[8];
  float ssq = 0.f;
  #pragma unroll
  for (int j = 0; j < 8; ++j) {
    x[j] = in[base + j*256 + tid];
    ssq += x[j]*x[j];
  }
  ssq = blockReduceSum(ssq, sd);
  float n = fminf(fmaxf(sqrtf(ssq), EPSV), MAXNV);
  float s = artanh_c(n) / n;
  #pragma unroll
  for (int j = 0; j < 8; ++j) {
    float v = s * x[j];
    out_bf[base + j*256 + tid] = f2bf(v);
    if (out_f) out_f[base + j*256 + tid] = v;
  }
}

__global__ __launch_bounds__(256) void cvt_f32_bf16(
    const float* __restrict__ in, unsigned short* __restrict__ out, int n) {
  int i = blockIdx.x * 256 + threadIdx.x;
  if (i < n) out[i] = f2bf(in[i]);
}

// ---------------------------------------------------------------------------
// GEMM: C[M,N] = A[M,K] @ W[N,K]^T (+ bias[N]), bf16 in, f32 out.
// Block tile 128x128, BK=32, double-buffered LDS (row stride 40 u16 = 80 B).
// 8 waves = 2(M) x 4(N); each wave: 4x2 tiles of v_wmma_f32_16x16x32_bf16.
// ---------------------------------------------------------------------------
#define BM 128
#define BN 128
#define BK 32
#define LDSW 40

union FragU { v16bf v; uint4 q[2]; };

#if USE_ASYNC_LDS
typedef __attribute__((address_space(1))) v4i* gv4i_p;
typedef __attribute__((address_space(3))) v4i* lv4i_p;
__device__ __forceinline__ void async_cp16(const unsigned short* g, unsigned short* l) {
  __builtin_amdgcn_global_load_async_to_lds_b128((gv4i_p)g, (lv4i_p)l, 0, 0);
}
#endif

__global__ __launch_bounds__(256) void gemm_bf16_nt(
    const unsigned short* __restrict__ A,
    const unsigned short* __restrict__ W,
    const float* __restrict__ bias,
    float* __restrict__ C, int M, int N, int K) {
  __shared__ unsigned short As[2][BM * LDSW];
  __shared__ unsigned short Bs[2][BN * LDSW];

  const int tid  = threadIdx.x;
  const int lane = tid & 31;
  const int wave = tid >> 5;
  const int wm   = wave >> 2;   // 0..1
  const int wn   = wave & 3;    // 0..3
  const int l15  = lane & 15;
  const int lhi  = lane >> 4;

  const int gm0 = blockIdx.y * BM;
  const int gn0 = blockIdx.x * BN;

  // two 16-byte chunks per tile per thread: chunk -> (row, kchunk)
  const int c0  = tid * 2;
  const int ar0 = c0 >> 2,       ak0 = (c0 & 3) * 8;
  const int ar1 = (c0 + 1) >> 2, ak1 = ((c0 + 1) & 3) * 8;

  const int nsteps = K / BK;

  const v8f vzero = {0.f,0.f,0.f,0.f,0.f,0.f,0.f,0.f};
  v8f acc[4][2];
  #pragma unroll
  for (int i = 0; i < 4; ++i)
    #pragma unroll
    for (int j = 0; j < 2; ++j) acc[i][j] = vzero;

#if USE_ASYNC_LDS
  // ---- async global->LDS pipeline (ASYNCcnt) ----
  {
    async_cp16(A + (size_t)(gm0 + ar0) * K + ak0, &As[0][ar0 * LDSW + ak0]);
    async_cp16(A + (size_t)(gm0 + ar1) * K + ak1, &As[0][ar1 * LDSW + ak1]);
    async_cp16(W + (size_t)(gn0 + ar0) * K + ak0, &Bs[0][ar0 * LDSW + ak0]);
    async_cp16(W + (size_t)(gn0 + ar1) * K + ak1, &Bs[0][ar1 * LDSW + ak1]);
  }
  for (int ks = 0; ks < nsteps; ++ks) {
    const int cur = ks & 1;
    const int nxt = cur ^ 1;
    if (ks + 1 < nsteps) {
      const size_t kk = (size_t)(ks + 1) * BK;
      async_cp16(A + (size_t)(gm0 + ar0) * K + kk + ak0, &As[nxt][ar0 * LDSW + ak0]);
      async_cp16(A + (size_t)(gm0 + ar1) * K + kk + ak1, &As[nxt][ar1 * LDSW + ak1]);
      async_cp16(W + (size_t)(gn0 + ar0) * K + kk + ak0, &Bs[nxt][ar0 * LDSW + ak0]);
      async_cp16(W + (size_t)(gn0 + ar1) * K + kk + ak1, &Bs[nxt][ar1 * LDSW + ak1]);
      __builtin_amdgcn_s_wait_asynccnt(4);   // current tile's 4 asyncs retired
    } else {
      __builtin_amdgcn_s_wait_asynccnt(0);
    }
    __syncthreads();                          // all waves' tile data visible

    FragU fa[4];
    #pragma unroll
    for (int i = 0; i < 4; ++i) {
      const unsigned short* p = &As[cur][(wm*64 + i*16 + l15) * LDSW + lhi*8];
      fa[i].q[0] = *(const uint4*)(p);
      fa[i].q[1] = *(const uint4*)(p + 16);
    }
    FragU fb[2];
    #pragma unroll
    for (int j = 0; j < 2; ++j) {
      const unsigned short* p = &Bs[cur][(wn*32 + j*16 + l15) * LDSW + lhi*8];
      fb[j].q[0] = *(const uint4*)(p);
      fb[j].q[1] = *(const uint4*)(p + 16);
    }
    #pragma unroll
    for (int i = 0; i < 4; ++i)
      #pragma unroll
      for (int j = 0; j < 2; ++j)
        acc[i][j] = __builtin_amdgcn_wmma_f32_16x16x32_bf16(
            false, fa[i].v, false, fb[j].v, (short)0, acc[i][j], false, false);

    __syncthreads();   // reads done before next iter's asyncs overwrite
  }
#else
  // ---- fallback: global->VGPR->LDS staging ----
  uint4 pa0, pa1, pb0, pb1;
  pa0 = *(const uint4*)(A + (size_t)(gm0 + ar0) * K + ak0);
  pa1 = *(const uint4*)(A + (size_t)(gm0 + ar1) * K + ak1);
  pb0 = *(const uint4*)(W + (size_t)(gn0 + ar0) * K + ak0);
  pb1 = *(const uint4*)(W + (size_t)(gn0 + ar1) * K + ak1);
  *(uint4*)&As[0][ar0 * LDSW + ak0] = pa0;
  *(uint4*)&As[0][ar1 * LDSW + ak1] = pa1;
  *(uint4*)&Bs[0][ar0 * LDSW + ak0] = pb0;
  *(uint4*)&Bs[0][ar1 * LDSW + ak1] = pb1;
  __syncthreads();

  for (int ks = 0; ks < nsteps; ++ks) {
    const int cur = ks & 1;
    const int nxt = cur ^ 1;
    if (ks + 1 < nsteps) {
      const size_t kk = (size_t)(ks + 1) * BK;
      pa0 = *(const uint4*)(A + (size_t)(gm0 + ar0) * K + kk + ak0);
      pa1 = *(const uint4*)(A + (size_t)(gm0 + ar1) * K + kk + ak1);
      pb0 = *(const uint4*)(W + (size_t)(gn0 + ar0) * K + kk + ak0);
      pb1 = *(const uint4*)(W + (size_t)(gn0 + ar1) * K + kk + ak1);
    }
    FragU fa[4];
    #pragma unroll
    for (int i = 0; i < 4; ++i) {
      const unsigned short* p = &As[cur][(wm*64 + i*16 + l15) * LDSW + lhi*8];
      fa[i].q[0] = *(const uint4*)(p);
      fa[i].q[1] = *(const uint4*)(p + 16);
    }
    FragU fb[2];
    #pragma unroll
    for (int j = 0; j < 2; ++j) {
      const unsigned short* p = &Bs[cur][(wn*32 + j*16 + l15) * LDSW + lhi*8];
      fb[j].q[0] = *(const uint4*)(p);
      fb[j].q[1] = *(const uint4*)(p + 16);
    }
    #pragma unroll
    for (int i = 0; i < 4; ++i)
      #pragma unroll
      for (int j = 0; j < 2; ++j)
        acc[i][j] = __builtin_amdgcn_wmma_f32_16x16x32_bf16(
            false, fa[i].v, false, fb[j].v, (short)0, acc[i][j], false, false);

    if (ks + 1 < nsteps) {
      *(uint4*)&As[nxt][ar0 * LDSW + ak0] = pa0;
      *(uint4*)&As[nxt][ar1 * LDSW + ak1] = pa1;
      *(uint4*)&Bs[nxt][ar0 * LDSW + ak0] = pb0;
      *(uint4*)&Bs[nxt][ar1 * LDSW + ak1] = pb1;
    }
    __syncthreads();
  }
#endif

  // Epilogue: C/D layout — VGPR v, lane l: M = v + 8*(l>>4), N = l&15
  float bv[2];
  #pragma unroll
  for (int j = 0; j < 2; ++j) {
    const int n = gn0 + wn*32 + j*16 + l15;
    bv[j] = bias ? bias[n] : 0.f;
  }
  #pragma unroll
  for (int i = 0; i < 4; ++i)
    #pragma unroll
    for (int j = 0; j < 2; ++j) {
      const int n = gn0 + wn*32 + j*16 + l15;
      #pragma unroll
      for (int v = 0; v < 8; ++v) {
        const int m = gm0 + wm*64 + i*16 + v + 8*lhi;
        C[(size_t)m * N + n] = acc[i][j][v] + bv[j];
      }
    }
}

// ---------------------------------------------------------------------------
__global__ __launch_bounds__(256) void gate_kernel(
    const float* __restrict__ lr, const float* __restrict__ ur,
    const float* __restrict__ lz, const float* __restrict__ uz,
    const float* __restrict__ loghx,
    float* __restrict__ zout, unsigned short* __restrict__ vbf) {
  __shared__ float sd[8];
  const int tid = threadIdx.x;
  const size_t base = (size_t)blockIdx.x * DIM;

  float a[8], b[8], c[8], d[8], h[8];
  #pragma unroll
  for (int j = 0; j < 8; ++j) {
    const size_t i = base + j*256 + tid;
    a[j] = lr[i]; b[j] = ur[i]; c[j] = lz[i]; d[j] = uz[i]; h[j] = loghx[i];
  }

  zero_exp8(a, sd); zero_exp8(b, sd);
  mobius8(a, b, sd);
  zero_log8(a, sd);
  #pragma unroll
  for (int j = 0; j < 8; ++j) a[j] = 1.f / (1.f + expf(-a[j]));   // r

  zero_exp8(c, sd); zero_exp8(d, sd);
  mobius8(c, d, sd);
  zero_log8(c, sd);
  #pragma unroll
  for (int j = 0; j < 8; ++j) c[j] = 1.f / (1.f + expf(-c[j]));   // z

  #pragma unroll
  for (int j = 0; j < 8; ++j) {
    const size_t i = base + j*256 + tid;
    zout[i] = c[j];
    vbf[i]  = f2bf(a[j] * h[j]);
  }
}

// ---------------------------------------------------------------------------
__global__ __launch_bounds__(256) void final_kernel(
    const float* __restrict__ whp, const float* __restrict__ uup,
    const float* __restrict__ hx, const float* __restrict__ z,
    float* __restrict__ out) {
  __shared__ float sd[8];
  const int tid = threadIdx.x;
  const size_t base = (size_t)blockIdx.x * DIM;

  float wh[8], ue[8], h[8], zz[8];
  #pragma unroll
  for (int j = 0; j < 8; ++j) {
    const size_t i = base + j*256 + tid;
    wh[j] = whp[i]; ue[j] = uup[i]; h[j] = hx[i]; zz[j] = z[i];
  }

  zero_exp8(wh, sd);
  zero_exp8(ue, sd);
  mobius8(wh, ue, sd);
  zero_log8(wh, sd);
  #pragma unroll
  for (int j = 0; j < 8; ++j) wh[j] = tanhf(wh[j]);
  zero_exp8(wh, sd);                       // inter

  float nh[8];
  #pragma unroll
  for (int j = 0; j < 8; ++j) nh[j] = -h[j];
  mobius8(nh, wh, sd);                     // mobius(-hx, inter)
  zero_log8(nh, sd);
  #pragma unroll
  for (int j = 0; j < 8; ++j) nh[j] *= zz[j];
  zero_exp8(nh, sd);                       // e

  mobius8(h, nh, sd);                      // out = mobius(hx, e)
  #pragma unroll
  for (int j = 0; j < 8; ++j) out[base + j*256 + tid] = h[j];
}

// ---------------------------------------------------------------------------
extern "C" void kernel_launch(void* const* d_in, const int* in_sizes, int n_in,
                              void* d_out, int out_size, void* d_ws, size_t ws_size,
                              hipStream_t stream) {
  const float* x     = (const float*)d_in[0];
  const float* hx    = (const float*)d_in[1];
  const float* w_r   = (const float*)d_in[2];
  const float* u_r_w = (const float*)d_in[3];
  const float* u_r_b = (const float*)d_in[4];
  const float* w_z   = (const float*)d_in[5];
  const float* u_z_w = (const float*)d_in[6];
  const float* u_z_b = (const float*)d_in[7];
  const float* u_w   = (const float*)d_in[8];
  const float* u_b   = (const float*)d_in[9];
  const float* w     = (const float*)d_in[10];
  float* out = (float*)d_out;

  const size_t BD = (size_t)BROWS * DIM;
  const size_t DD = (size_t)DIM * DIM;

  char* ws = (char*)d_ws;
  unsigned short* logx_bf  = (unsigned short*)ws;            ws += BD * 2;
  unsigned short* loghx_bf = (unsigned short*)ws;            ws += BD * 2;
  float*          loghx_f  = (float*)ws;                     ws += BD * 4;
  unsigned short* wr_bf    = (unsigned short*)ws;            ws += DD * 2;
  unsigned short* urw_bf   = (unsigned short*)ws;            ws += DD * 2;
  unsigned short* wz_bf    = (unsigned short*)ws;            ws += DD * 2;
  unsigned short* uzw_bf   = (unsigned short*)ws;            ws += DD * 2;
  unsigned short* uw_bf    = (unsigned short*)ws;            ws += DD * 2;
  unsigned short* w_bf     = (unsigned short*)ws;            ws += DD * 2;
  float* lr = (float*)ws;                                    ws += BD * 4;
  float* ur = (float*)ws;                                    ws += BD * 4;
  float* lz = (float*)ws;                                    ws += BD * 4;
  float* uz = (float*)ws;                                    ws += BD * 4;
  float* uu = (float*)ws;                                    ws += BD * 4;

  // Aliased buffers (safe via stream ordering / read-before-write within kernels)
  float* zbuf = lr;
  unsigned short* vbf = logx_bf;
  float* whp = ur;

  zero_log_prep<<<BROWS, 256, 0, stream>>>(x,  logx_bf,  nullptr);
  zero_log_prep<<<BROWS, 256, 0, stream>>>(hx, loghx_bf, loghx_f);

  const int nW = (int)DD, gW = (nW + 255) / 256;
  cvt_f32_bf16<<<gW, 256, 0, stream>>>(w_r,   wr_bf,  nW);
  cvt_f32_bf16<<<gW, 256, 0, stream>>>(u_r_w, urw_bf, nW);
  cvt_f32_bf16<<<gW, 256, 0, stream>>>(w_z,   wz_bf,  nW);
  cvt_f32_bf16<<<gW, 256, 0, stream>>>(u_z_w, uzw_bf, nW);
  cvt_f32_bf16<<<gW, 256, 0, stream>>>(u_w,   uw_bf,  nW);
  cvt_f32_bf16<<<gW, 256, 0, stream>>>(w,     w_bf,   nW);

  dim3 grid(DIM / BN, BROWS / BM), block(256);
  gemm_bf16_nt<<<grid, block, 0, stream>>>(loghx_bf, wr_bf,  nullptr, lr, BROWS, DIM, DIM);
  gemm_bf16_nt<<<grid, block, 0, stream>>>(logx_bf,  urw_bf, u_r_b,   ur, BROWS, DIM, DIM);
  gemm_bf16_nt<<<grid, block, 0, stream>>>(loghx_bf, wz_bf,  nullptr, lz, BROWS, DIM, DIM);
  gemm_bf16_nt<<<grid, block, 0, stream>>>(logx_bf,  uzw_bf, u_z_b,   uz, BROWS, DIM, DIM);
  gemm_bf16_nt<<<grid, block, 0, stream>>>(logx_bf,  uw_bf,  u_b,     uu, BROWS, DIM, DIM);

  gate_kernel<<<BROWS, 256, 0, stream>>>(lr, ur, lz, uz, loghx_f, zbuf, vbf);

  gemm_bf16_nt<<<grid, block, 0, stream>>>(vbf, w_bf, nullptr, whp, BROWS, DIM, DIM);

  final_kernel<<<BROWS, 256, 0, stream>>>(whp, uu, hx, zbuf, out);
}
